// MultiHeadLatentAttention_15874199126583
// MI455X (gfx1250) — matrix operand order repaired and checked by reading, not compiled, not run on hardware
//
#include <hip/hip_runtime.h>
#include <hip/hip_bf16.h>
#include <math.h>

#define DIM    2048
#define H      16
#define C_KV   512
#define C_Q    1536
#define D_NOPE 128
#define D_ROPE 64
#define D_V    128
#define BATCH  2
#define SEQ    2048
#define QH     (D_NOPE + D_ROPE)   // 192
#define KVH    (D_NOPE + D_V)      // 256
#define ROWS   (BATCH * SEQ)       // 4096

typedef __attribute__((ext_vector_type(16))) __bf16 bf16x16;
typedef __attribute__((ext_vector_type(8)))  __bf16 bf16x8;
typedef __attribute__((ext_vector_type(8)))  float  f32x8;

__device__ __forceinline__ bf16x16 join8(bf16x8 lo, bf16x8 hi) {
    return __builtin_shufflevector(lo, hi, 0,1,2,3,4,5,6,7,8,9,10,11,12,13,14,15);
}
// 16 contiguous bf16 (two aligned 16B halves) -> B-operand fragment
__device__ __forceinline__ bf16x16 ld_b16x16(const __bf16* p) {
    return join8(*(const bf16x8*)p, *(const bf16x8*)(p + 8));
}
// A-operand fragment: lane holds K run [half*16, half*16+16)
__device__ __forceinline__ bf16x16 ld_afrag(const __bf16* row, int kbase, int half) {
    bf16x8 lo = *(const bf16x8*)(row + kbase + half * 16);
    bf16x8 hi = *(const bf16x8*)(row + kbase + half * 16 + 8);
    return join8(lo, hi);
}
__device__ __forceinline__ f32x8 wmma_bf16(bf16x16 a, bf16x16 b, f32x8 c) {
    return __builtin_amdgcn_wmma_f32_16x16x32_bf16(false, a, false, b, (short)0, c, false, false);
}

// ---------------------------------------------------------------------------
// Generic GEMM: C(MxN,f32) = A(MxK,f32,row) @ B(KxN,f32,row); bf16 WMMA math.
// Block = 128 threads (4 waves). Block tile 128(M) x 64(N), K step 64.
// Wave w owns rows [w*32, w*32+32): 2 M sub-tiles x 4 N sub-tiles = 8 accums.
// Per K-step: 16 WMMA vs 24 ds_load_b128; B frags reused across 2 A frags.
// ---------------------------------------------------------------------------
__global__ __launch_bounds__(128)
void gemm_f32in_bf16wmma(const float* __restrict__ A, const float* __restrict__ Bm,
                         float* __restrict__ C, int M, int N, int K) {
    __shared__ __attribute__((aligned(16))) __bf16 As[128][72]; // [m][k], 144B pitch
    __shared__ __attribute__((aligned(16))) __bf16 Bs[64][72];  // [n][k] (transposed)
    const int t     = threadIdx.x;
    const int w     = t >> 5;
    const int lane  = t & 31;
    const int col16 = lane & 15;
    const int half  = lane >> 4;
    const int mBase = blockIdx.y * 128;
    const int nBase = blockIdx.x * 64;

    f32x8 acc[2][4] = {};
    for (int k0 = 0; k0 < K; k0 += 64) {
        __syncthreads();
        for (int idx = t; idx < 128 * 64; idx += 128) {
            int r = idx >> 6, c = idx & 63;
            As[r][c] = (__bf16)A[(size_t)(mBase + r) * K + (k0 + c)];
        }
        for (int idx = t; idx < 64 * 64; idx += 128) {
            int r = idx >> 6, c = idx & 63;
            Bs[c][r] = (__bf16)Bm[(size_t)(k0 + r) * N + (nBase + c)];
        }
        __syncthreads();
        if (k0 + 64 < K) {  // gfx1250 global_prefetch_b8 hints for next K tile
            __builtin_prefetch(&A[(size_t)(mBase + t) * K + k0 + 64], 0, 1);
            __builtin_prefetch(&Bm[(size_t)(k0 + 64 + (t & 63)) * N + nBase + (t >> 6) * 32], 0, 1);
        }
#pragma unroll
        for (int kc = 0; kc < 2; ++kc) {
            // batch all fragment loads, then burst the WMMAs
            bf16x16 a0 = ld_afrag(&As[w * 32 + col16][0],      kc * 32, half);
            bf16x16 a1 = ld_afrag(&As[w * 32 + 16 + col16][0], kc * 32, half);
            bf16x16 bfr[4];
#pragma unroll
            for (int nt = 0; nt < 4; ++nt)
                bfr[nt] = ld_b16x16(&Bs[nt * 16 + col16][kc * 32 + half * 16]);
#pragma unroll
            for (int nt = 0; nt < 4; ++nt) {
                acc[0][nt] = wmma_bf16(a0, bfr[nt], acc[0][nt]);
                acc[1][nt] = wmma_bf16(a1, bfr[nt], acc[1][nt]);
            }
        }
    }
#pragma unroll
    for (int mi = 0; mi < 2; ++mi)
#pragma unroll
        for (int nt = 0; nt < 4; ++nt)
#pragma unroll
            for (int r = 0; r < 8; ++r) {
                int row = mBase + w * 32 + mi * 16 + half * 8 + r;
                int cc  = nBase + nt * 16 + col16;
                C[(size_t)row * N + cc] = acc[mi][nt][r];
            }
}

// ---------------------------------------------------------------------------
// In-place RMSNorm, one block per row.
// ---------------------------------------------------------------------------
__global__ __launch_bounds__(256)
void rmsnorm_inplace(float* __restrict__ x, const float* __restrict__ scale, int n) {
    float* row = x + (size_t)blockIdx.x * n;
    float ss = 0.f;
    for (int i = threadIdx.x; i < n; i += 256) { float v = row[i]; ss += v * v; }
#pragma unroll
    for (int off = 16; off > 0; off >>= 1) ss += __shfl_down(ss, off, 32);
    __shared__ float red[8];
    if ((threadIdx.x & 31) == 0) red[threadIdx.x >> 5] = ss;
    __syncthreads();
    if (threadIdx.x == 0) {
        float tot = 0.f;
        for (int i = 0; i < 8; ++i) tot += red[i];
        red[0] = rsqrtf(tot / (float)n + 1e-6f);
    }
    __syncthreads();
    float inv = red[0];
    for (int i = threadIdx.x; i < n; i += 256) row[i] = row[i] * inv * scale[i];
}

// ---------------------------------------------------------------------------
// RoPE on q_pe (last 64 of each head of q) and k_pe. Block per (b,s) row;
// groups 0..H-1 -> q heads, group H -> k_pe.
// ---------------------------------------------------------------------------
__global__ __launch_bounds__((H + 1) * 32)
void rope_kernel(float* __restrict__ q, float* __restrict__ kpe) {
    int rowIdx = blockIdx.x;
    int pos = rowIdx & (SEQ - 1);
    int g = threadIdx.x >> 5;
    int j = threadIdx.x & 31;
    float freq = __expf(-((float)(2 * j) / 64.f) * 9.210340372f);  // 10000^(-2j/64)
    float ang = (float)pos * freq;
    float c = __cosf(ang), s = __sinf(ang);
    float *p1, *p2;
    if (g < H) {
        float* base = q + ((size_t)rowIdx * H + g) * QH + D_NOPE;
        p1 = base + j; p2 = base + 32 + j;
    } else {
        float* base = kpe + (size_t)rowIdx * D_ROPE;
        p1 = base + j; p2 = base + 32 + j;
    }
    float x1 = *p1, x2 = *p2;
    *p1 = x1 * c - x2 * s;
    *p2 = x2 * c + x1 * s;
}

// ---------------------------------------------------------------------------
// Causal flash attention. Grid (S/64, H, B); 128 threads; wave w owns 16 q rows.
// Q eff dim 192 = k_nope(128) ++ k_pe(64, head-broadcast); V dim 128.
// ---------------------------------------------------------------------------
__global__ __launch_bounds__(128)
void mla_flash_attn(const float* __restrict__ q, const float* __restrict__ kv,
                    const float* __restrict__ kpe, float* __restrict__ out) {
    __shared__ __attribute__((aligned(16))) __bf16 Ks[64][200];   // [key][192+pad]
    __shared__ __attribute__((aligned(16))) __bf16 Vt[128][72];   // [d][key+pad]
    __shared__ __attribute__((aligned(16))) __bf16 Ps[4][16][72]; // per-wave P
    const int t     = threadIdx.x;
    const int w     = t >> 5;
    const int lane  = t & 31;
    const int col16 = lane & 15;
    const int half  = lane >> 4;
    const int h     = blockIdx.y;
    const int qb    = blockIdx.x * 64;
    const size_t bS = (size_t)blockIdx.z * SEQ;

    bf16x16 qf[6];
    {
        const float* qrow = q + ((bS + qb + w * 16 + col16) * H + h) * (size_t)QH;
#pragma unroll
        for (int kc = 0; kc < 6; ++kc) {
            bf16x16 a;
#pragma unroll
            for (int i = 0; i < 8; ++i) {
                a[i]     = (__bf16)qrow[kc * 32 + half * 16 + i];
                a[i + 8] = (__bf16)qrow[kc * 32 + half * 16 + 8 + i];
            }
            qf[kc] = a;
        }
    }

    f32x8 o[8] = {};
    float mPrev[8], lPrev[8];
#pragma unroll
    for (int r = 0; r < 8; ++r) { mPrev[r] = -INFINITY; lPrev[r] = 0.f; }

    const float scale = 0.07216878364870322f;  // 1/sqrt(192)
    const int nTiles = (qb >> 6) + 1;          // causal: keys <= query
    for (int kt = 0; kt < nTiles; ++kt) {
        const int kb = kt * 64;
        __syncthreads();
        for (int idx = t; idx < 64 * 192; idx += 128) {
            int r = idx / 192, c = idx - r * 192;
            size_t grow = bS + kb + r;
            float v = (c < D_NOPE) ? kv[(grow * H + h) * KVH + c]
                                   : kpe[grow * D_ROPE + (c - D_NOPE)];
            Ks[r][c] = (__bf16)v;
        }
        for (int idx = t; idx < 64 * 128; idx += 128) {
            int r = idx >> 7, d = idx & 127;
            size_t grow = bS + kb + r;
            Vt[d][r] = (__bf16)kv[(grow * H + h) * KVH + D_NOPE + d];
        }
        __syncthreads();

        f32x8 sAcc[4] = {};
#pragma unroll
        for (int kc = 0; kc < 6; ++kc) {
            bf16x16 bfr[4];
#pragma unroll
            for (int nt = 0; nt < 4; ++nt)
                bfr[nt] = ld_b16x16(&Ks[nt * 16 + col16][kc * 32 + half * 16]);
#pragma unroll
            for (int nt = 0; nt < 4; ++nt)
                sAcc[nt] = wmma_bf16(qf[kc], bfr[nt], sAcc[nt]);
        }

        const bool diag = (kt == nTiles - 1);
#pragma unroll
        for (int nt = 0; nt < 4; ++nt) {
            int gcol = kb + nt * 16 + col16;
#pragma unroll
            for (int r = 0; r < 8; ++r) {
                float sv = sAcc[nt][r] * scale;
                if (diag && (gcol > qb + w * 16 + half * 8 + r)) sv = -1e30f;
                sAcc[nt][r] = sv;
            }
        }

        float mLoc[8];
#pragma unroll
        for (int r = 0; r < 8; ++r)
            mLoc[r] = fmaxf(fmaxf(sAcc[0][r], sAcc[1][r]), fmaxf(sAcc[2][r], sAcc[3][r]));
#pragma unroll
        for (int off = 1; off < 16; off <<= 1)
#pragma unroll
            for (int r = 0; r < 8; ++r)
                mLoc[r] = fmaxf(mLoc[r], __shfl_xor(mLoc[r], off, 32));

        float alpha[8], lLoc[8], mNew[8];
#pragma unroll
        for (int r = 0; r < 8; ++r) {
            mNew[r]  = fmaxf(mPrev[r], mLoc[r]);
            alpha[r] = __expf(mPrev[r] - mNew[r]);
            lLoc[r]  = 0.f;
        }
#pragma unroll
        for (int nt = 0; nt < 4; ++nt)
#pragma unroll
            for (int r = 0; r < 8; ++r) {
                float p = __expf(sAcc[nt][r] - mNew[r]);
                sAcc[nt][r] = p;
                lLoc[r] += p;
            }
#pragma unroll
        for (int off = 1; off < 16; off <<= 1)
#pragma unroll
            for (int r = 0; r < 8; ++r)
                lLoc[r] += __shfl_xor(lLoc[r], off, 32);
#pragma unroll
        for (int r = 0; r < 8; ++r) {
            lPrev[r] = lPrev[r] * alpha[r] + lLoc[r];
            mPrev[r] = mNew[r];
        }
#pragma unroll
        for (int dt = 0; dt < 8; ++dt)
#pragma unroll
            for (int r = 0; r < 8; ++r) o[dt][r] *= alpha[r];

        // C-layout -> A-layout transpose of P through LDS
#pragma unroll
        for (int nt = 0; nt < 4; ++nt)
#pragma unroll
            for (int r = 0; r < 8; ++r)
                Ps[w][half * 8 + r][nt * 16 + col16] = (__bf16)sAcc[nt][r];
        __syncthreads();

#pragma unroll
        for (int kc = 0; kc < 2; ++kc) {
            bf16x16 a = ld_afrag(&Ps[w][col16][0], kc * 32, half);
#pragma unroll
            for (int db = 0; db < 2; ++db) {
                bf16x16 bfr[4];
#pragma unroll
                for (int j = 0; j < 4; ++j)
                    bfr[j] = ld_b16x16(&Vt[(db * 4 + j) * 16 + col16][kc * 32 + half * 16]);
#pragma unroll
                for (int j = 0; j < 4; ++j)
                    o[db * 4 + j] = wmma_bf16(a, bfr[j], o[db * 4 + j]);
            }
        }
    }

#pragma unroll
    for (int r = 0; r < 8; ++r) lPrev[r] = 1.f / lPrev[r];
#pragma unroll
    for (int dt = 0; dt < 8; ++dt)
#pragma unroll
        for (int r = 0; r < 8; ++r) {
            size_t grow = bS + qb + w * 16 + half * 8 + r;
            out[(grow * H + h) * (size_t)D_V + dt * 16 + col16] = o[dt][r] * lPrev[r];
        }
}

// ---------------------------------------------------------------------------
extern "C" void kernel_launch(void* const* d_in, const int* in_sizes, int n_in,
                              void* d_out, int out_size, void* d_ws, size_t ws_size,
                              hipStream_t stream) {
    (void)in_sizes; (void)n_in; (void)out_size; (void)ws_size;
    const float* x        = (const float*)d_in[0];
    const float* w_dq     = (const float*)d_in[1];
    const float* q_scale  = (const float*)d_in[2];
    const float* w_uq     = (const float*)d_in[3];
    const float* w_dkv    = (const float*)d_in[4];
    const float* kv_scale = (const float*)d_in[5];
    const float* w_ukv    = (const float*)d_in[6];
    const float* w_kpe    = (const float*)d_in[7];
    const float* w_o      = (const float*)d_in[8];
    float* out = (float*)d_out;

    float* ws     = (float*)d_ws;
    float* q_lat  = ws;                                   // ROWS * C_Q
    float* qbuf   = q_lat  + (size_t)ROWS * C_Q;          // ROWS * H*QH
    float* kv_lat = qbuf   + (size_t)ROWS * H * QH;       // ROWS * C_KV
    float* kvbuf  = kv_lat + (size_t)ROWS * C_KV;         // ROWS * H*KVH
    float* kpe    = kvbuf  + (size_t)ROWS * H * KVH;      // ROWS * D_ROPE
    float* attn   = kpe    + (size_t)ROWS * D_ROPE;       // ROWS * H*D_V

    auto gemm = [&](const float* A, const float* B, float* C, int M, int N, int K) {
        dim3 g(N / 64, M / 128);
        hipLaunchKernelGGL(gemm_f32in_bf16wmma, g, dim3(128), 0, stream, A, B, C, M, N, K);
    };

    gemm(x, w_dq, q_lat, ROWS, C_Q, DIM);
    hipLaunchKernelGGL(rmsnorm_inplace, dim3(ROWS), dim3(256), 0, stream, q_lat, q_scale, C_Q);
    gemm(q_lat, w_uq, qbuf, ROWS, H * QH, C_Q);
    gemm(x, w_dkv, kv_lat, ROWS, C_KV, DIM);
    hipLaunchKernelGGL(rmsnorm_inplace, dim3(ROWS), dim3(256), 0, stream, kv_lat, kv_scale, C_KV);
    gemm(kv_lat, w_ukv, kvbuf, ROWS, H * KVH, C_KV);
    gemm(x, w_kpe, kpe, ROWS, D_ROPE, DIM);
    hipLaunchKernelGGL(rope_kernel, dim3(ROWS), dim3((H + 1) * 32), 0, stream, qbuf, kpe);
    hipLaunchKernelGGL(mla_flash_attn, dim3(SEQ / 64, H, BATCH), dim3(128), 0, stream,
                       qbuf, kvbuf, kpe, attn);
    gemm(attn, w_o, out, ROWS, DIM, H * D_V);
}
